// LSTMCell_17480516895034
// MI455X (gfx1250) — matrix operand order repaired and checked by reading, not compiled, not run on hardware
//
#include <hip/hip_runtime.h>

// MI455X / gfx1250, wave32. All matrix ops via v_wmma_f32_16x16x32_f16.
#define HN  2048
#define HN2 ((size_t)HN * (size_t)HN)

typedef __attribute__((ext_vector_type(16))) _Float16 v16h;
typedef __attribute__((ext_vector_type(8)))  _Float16 v8h;
typedef __attribute__((ext_vector_type(8)))  float    v8f;

__device__ __forceinline__ v16h cat8(v8h lo, v8h hi) {
  return __builtin_shufflevector(lo, hi, 0,1,2,3,4,5,6,7,8,9,10,11,12,13,14,15);
}

// ---------------------------------------------------------------------------
// C = A1*B1^T (+ A2*B2^T) (+ bias), M=N=K=2048, A row-major f16, B stored
// pre-transposed [N,K] f16 so B-lane register slices are contiguous loads.
// Block = 256 threads = 8 waves; block tile 128x128; wave tile 32x64
// (2 M-subtiles x 4 N-subtiles -> 8 WMMA per 32-wide K step).
// ---------------------------------------------------------------------------
struct Gemm2Args {
  const _Float16* A1[4];
  const _Float16* B1[4];
  const _Float16* A2[4];
  const _Float16* B2[4];
  const float*    bias[4];
  float*          C[4];
};

__global__ __launch_bounds__(256)
void gemm2_wmma_kernel(Gemm2Args args) {
  const int z    = blockIdx.z;
  const int lane = threadIdx.x & 31;
  const int wave = threadIdx.x >> 5;
  const int wm   = wave & 3;                 // 4 waves along M
  const int wn   = wave >> 2;                // 2 waves along N
  const int m0   = blockIdx.x * 128 + wm * 32;
  const int n0   = blockIdx.y * 128 + wn * 64;
  const int r16  = lane & 15;                // row-in-tile / col-in-tile
  const int kh   = lane >> 4;                // which K-half this lane owns

  v8f acc[2][4] = {};

  const _Float16* Alist[2] = { args.A1[z], args.A2[z] };
  const _Float16* Blist[2] = { args.B1[z], args.B2[z] };

  for (int p = 0; p < 2; ++p) {
    const _Float16* A = Alist[p];
    const _Float16* B = Blist[p];
    if (A == nullptr) continue;              // wave-uniform branch
    // A: lane holds row (m+r16), K = kh*16 + [0..15] (matches B order)
    const _Float16* a0  = A + (size_t)(m0 + r16) * HN + kh * 16;
    const _Float16* a1  = a0 + (size_t)16 * HN;
    // B^T: lane holds col (n+r16), K = kh*16 + [0..15] (contiguous K, same order as A)
    const _Float16* bp0 = B + (size_t)(n0 +  0 + r16) * HN + kh * 16;
    const _Float16* bp1 = B + (size_t)(n0 + 16 + r16) * HN + kh * 16;
    const _Float16* bp2 = B + (size_t)(n0 + 32 + r16) * HN + kh * 16;
    const _Float16* bp3 = B + (size_t)(n0 + 48 + r16) * HN + kh * 16;

    for (int k = 0; k < HN; k += 32) {
      v16h av0 = cat8(*(const v8h*)(a0),  *(const v8h*)(a0 + 8));
      v16h av1 = cat8(*(const v8h*)(a1),  *(const v8h*)(a1 + 8));
      v16h bv0 = cat8(*(const v8h*)(bp0), *(const v8h*)(bp0 + 8));
      v16h bv1 = cat8(*(const v8h*)(bp1), *(const v8h*)(bp1 + 8));
      v16h bv2 = cat8(*(const v8h*)(bp2), *(const v8h*)(bp2 + 8));
      v16h bv3 = cat8(*(const v8h*)(bp3), *(const v8h*)(bp3 + 8));

      acc[0][0] = __builtin_amdgcn_wmma_f32_16x16x32_f16(false, av0, false, bv0, (short)0, acc[0][0], false, false);
      acc[0][1] = __builtin_amdgcn_wmma_f32_16x16x32_f16(false, av0, false, bv1, (short)0, acc[0][1], false, false);
      acc[0][2] = __builtin_amdgcn_wmma_f32_16x16x32_f16(false, av0, false, bv2, (short)0, acc[0][2], false, false);
      acc[0][3] = __builtin_amdgcn_wmma_f32_16x16x32_f16(false, av0, false, bv3, (short)0, acc[0][3], false, false);
      acc[1][0] = __builtin_amdgcn_wmma_f32_16x16x32_f16(false, av1, false, bv0, (short)0, acc[1][0], false, false);
      acc[1][1] = __builtin_amdgcn_wmma_f32_16x16x32_f16(false, av1, false, bv1, (short)0, acc[1][1], false, false);
      acc[1][2] = __builtin_amdgcn_wmma_f32_16x16x32_f16(false, av1, false, bv2, (short)0, acc[1][2], false, false);
      acc[1][3] = __builtin_amdgcn_wmma_f32_16x16x32_f16(false, av1, false, bv3, (short)0, acc[1][3], false, false);

      a0 += 32; a1 += 32; bp0 += 32; bp1 += 32; bp2 += 32; bp3 += 32;
    }
  }

  // D layout: VGPR r -> row (kh*8 + r), col = r16 within each 16x16 subtile.
  const float* bias = args.bias[z];
  float* C = args.C[z];
  #pragma unroll
  for (int mt = 0; mt < 2; ++mt) {
    const int rowBase = m0 + mt * 16 + kh * 8;
    #pragma unroll
    for (int nt = 0; nt < 4; ++nt) {
      const int col = n0 + nt * 16 + r16;
      const float bv = bias ? bias[col] : 0.0f;
      #pragma unroll
      for (int r = 0; r < 8; ++r)
        C[(size_t)(rowBase + r) * HN + col] = acc[mt][nt][r] + bv;
    }
  }
}

// ---------------------------------------------------------------------------
// Row-wise LayerNorm + activation, fp32 in -> f16 row-major out.
// One 256-thread block per row; LDS tree reduction for mean / var.
// ---------------------------------------------------------------------------
struct LnArgs {
  const float* pre[4];
  const float* gamma[4];
  const float* beta[4];
  _Float16*    out[4];
  int          isTanh[4];
};

__global__ __launch_bounds__(256)
void ln_act_kernel(LnArgs a) {
  const int z    = blockIdx.y;
  const int rowi = blockIdx.x;
  const float* x = a.pre[z] + (size_t)rowi * HN;
  __shared__ float s1[256];
  __shared__ float s2[256];
  float v[8];
  float ls = 0.0f, lq = 0.0f;
  #pragma unroll
  for (int j = 0; j < 8; ++j) {
    v[j] = x[threadIdx.x + j * 256];
    ls += v[j];
    lq += v[j] * v[j];
  }
  s1[threadIdx.x] = ls;
  s2[threadIdx.x] = lq;
  __syncthreads();
  for (int s = 128; s > 0; s >>= 1) {
    if ((int)threadIdx.x < s) {
      s1[threadIdx.x] += s1[threadIdx.x + s];
      s2[threadIdx.x] += s2[threadIdx.x + s];
    }
    __syncthreads();
  }
  const float mean = s1[0] * (1.0f / HN);
  const float var  = s2[0] * (1.0f / HN) - mean * mean;
  const float rstd = rsqrtf(var + 1e-5f);
  const float* g = a.gamma[z];
  const float* b = a.beta[z];
  _Float16* o = a.out[z] + (size_t)rowi * HN;
  const int tnh = a.isTanh[z];
  #pragma unroll
  for (int j = 0; j < 8; ++j) {
    const int col = threadIdx.x + j * 256;
    float y = (v[j] - mean) * rstd * g[col] + b[col];
    y = tnh ? tanhf(y) : 1.0f / (1.0f + expf(-y));
    o[col] = (_Float16)y;
  }
}

// ---------------------------------------------------------------------------
// Tiled (32x32, LDS) transpose + convert to f16: dst[c*HN + r] = src[r*HN + c].
// Both global sides coalesced. Batched over blockIdx.z.
// ---------------------------------------------------------------------------
struct TrArgs {
  const void* src[10];
  _Float16*   dst[10];
};

template <typename SrcT>
__global__ __launch_bounds__(256)
void transpose_to_f16_kernel(TrArgs a) {
  const SrcT* src = (const SrcT*)a.src[blockIdx.z];
  _Float16*   dst = a.dst[blockIdx.z];
  __shared__ _Float16 tile[32][33];
  const int x  = threadIdx.x & 31;
  const int y  = threadIdx.x >> 5;     // 0..7
  const int r0 = blockIdx.x * 32;
  const int c0 = blockIdx.y * 32;
  #pragma unroll
  for (int j = 0; j < 4; ++j) {
    const int r = y + j * 8;
    tile[r][x] = (_Float16)src[(size_t)(r0 + r) * HN + c0 + x];
  }
  __syncthreads();
  #pragma unroll
  for (int j = 0; j < 4; ++j) {
    const int r = y + j * 8;
    dst[(size_t)(c0 + r) * HN + r0 + x] = tile[x][r];
  }
}

// ---------------------------------------------------------------------------
// Elementwise fp32 -> f16 (optional tanh). Batched over blockIdx.z.
// ---------------------------------------------------------------------------
struct CvArgs {
  const float* src[4];
  _Float16*    dst[4];
  int          doTanh[4];
};

__global__ __launch_bounds__(256)
void conv_f32_to_f16_kernel(CvArgs a) {
  const int z = blockIdx.z;
  const float* s = a.src[z];
  _Float16*    d = a.dst[z];
  const int tnh  = a.doTanh[z];
  const size_t base = ((size_t)blockIdx.x * 256 + threadIdx.x) * 8;
  #pragma unroll
  for (int j = 0; j < 8; ++j) {
    float v = s[base + j];
    if (tnh) v = tanhf(v);
    d[base + j] = (_Float16)v;
  }
}

// ---------------------------------------------------------------------------
extern "C" void kernel_launch(void* const* d_in, const int* in_sizes, int n_in,
                              void* d_out, int out_size, void* d_ws, size_t ws_size,
                              hipStream_t stream) {
  (void)in_sizes; (void)n_in; (void)out_size; (void)ws_size;

  const float* inp = (const float*)d_in[0];
  const float* cst = (const float*)d_in[1];
  const float* hst = (const float*)d_in[2];
  const float* Wh[4] = { (const float*)d_in[3], (const float*)d_in[4],
                         (const float*)d_in[5], (const float*)d_in[6] };
  const float* Wx[4] = { (const float*)d_in[7], (const float*)d_in[8],
                         (const float*)d_in[9], (const float*)d_in[10] };
  const float* bx[4] = { (const float*)d_in[11], (const float*)d_in[12],
                         (const float*)d_in[13], (const float*)d_in[14] };
  const float* Wy = (const float*)d_in[15];
  const float* by = (const float*)d_in[16];
  const float* gam[4] = { (const float*)d_in[17], (const float*)d_in[19],
                          (const float*)d_in[21], (const float*)d_in[23] };
  const float* bet[4] = { (const float*)d_in[18], (const float*)d_in[20],
                          (const float*)d_in[22], (const float*)d_in[24] };

  float* c_out = (float*)d_out;           // c_t [2048,2048]
  float* h_out = c_out + HN2;             // h_t
  float* y_out = h_out + HN2;             // y_t

  // ---- workspace carving (all 8/16 MB chunks, 16B-aligned trivially) ----
  char* p = (char*)d_ws;
  auto carve = [&](size_t bytes) { char* r = p; p += bytes; return r; };
  _Float16* x16  = (_Float16*)carve(HN2 * 2);
  _Float16* h16  = (_Float16*)carve(HN2 * 2);
  _Float16* gateRM[4];                    // f,g,i,o activations row-major f16
  for (int k = 0; k < 4; ++k) gateRM[k] = (_Float16*)carve(HN2 * 2);
  _Float16* i16t = (_Float16*)carve(HN2 * 2);
  _Float16* o16t = (_Float16*)carve(HN2 * 2);
  _Float16* c16t = (_Float16*)carve(HN2 * 2);   // c_states transposed
  _Float16* wht[4]; for (int k = 0; k < 4; ++k) wht[k] = (_Float16*)carve(HN2 * 2);
  _Float16* wxt[4]; for (int k = 0; k < 4; ++k) wxt[k] = (_Float16*)carve(HN2 * 2);
  _Float16* wyt     = (_Float16*)carve(HN2 * 2);
  _Float16* tanhc16 = (_Float16*)carve(HN2 * 2);
  _Float16* ht16    = (_Float16*)carve(HN2 * 2);
  float* pre[4];
  for (int k = 0; k < 4; ++k) pre[k] = (float*)carve(HN2 * 4);

  const dim3 blk(256);

  // 1) x, h -> f16 row-major (WMMA A operands)
  {
    CvArgs a = {};
    a.src[0] = inp; a.dst[0] = x16;
    a.src[1] = hst; a.dst[1] = h16;
    conv_f32_to_f16_kernel<<<dim3(HN2 / (256 * 8), 1, 2), blk, 0, stream>>>(a);
  }

  // 2) weights + c_states -> f16 transposed (WMMA B operands)
  {
    TrArgs t = {};
    for (int k = 0; k < 4; ++k) { t.src[k]     = Wh[k]; t.dst[k]     = wht[k]; }
    for (int k = 0; k < 4; ++k) { t.src[4 + k] = Wx[k]; t.dst[4 + k] = wxt[k]; }
    t.src[8] = Wy;  t.dst[8] = wyt;
    t.src[9] = cst; t.dst[9] = c16t;
    transpose_to_f16_kernel<float><<<dim3(64, 64, 10), blk, 0, stream>>>(t);
  }

  // 3) gate pre-activations: pre_k = h*W_hk^T? (stored [in,out], so B = W^T) + x*W_xk + b
  {
    Gemm2Args ga = {};
    for (int k = 0; k < 4; ++k) {
      ga.A1[k] = h16;  ga.B1[k] = wht[k];
      ga.A2[k] = x16;  ga.B2[k] = wxt[k];
      ga.bias[k] = bx[k];
      ga.C[k] = pre[k];
    }
    gemm2_wmma_kernel<<<dim3(16, 16, 4), blk, 0, stream>>>(ga);
  }

  // 4) LayerNorm + activation (f,i,o: sigmoid; g: tanh), write f16 row-major
  {
    LnArgs la = {};
    for (int k = 0; k < 4; ++k) {
      la.pre[k] = pre[k]; la.gamma[k] = gam[k]; la.beta[k] = bet[k];
      la.out[k] = gateRM[k]; la.isTanh[k] = 0;
    }
    la.isTanh[1] = 1;   // g gate uses tanh
    ln_act_kernel<<<dim3(HN, 4), blk, 0, stream>>>(la);
  }

  // 5) i_t, o_t -> transposed f16 (they appear as B operands downstream)
  {
    TrArgs t = {};
    t.src[0] = gateRM[2]; t.dst[0] = i16t;
    t.src[1] = gateRM[3]; t.dst[1] = o16t;
    transpose_to_f16_kernel<_Float16><<<dim3(64, 64, 2), blk, 0, stream>>>(t);
  }

  // 6) c_t = f_t @ c_states + g_t @ i_t  (fp32 out directly to d_out)
  {
    Gemm2Args ga = {};
    ga.A1[0] = gateRM[0]; ga.B1[0] = c16t;
    ga.A2[0] = gateRM[1]; ga.B2[0] = i16t;
    ga.bias[0] = nullptr; ga.C[0] = c_out;
    gemm2_wmma_kernel<<<dim3(16, 16, 1), blk, 0, stream>>>(ga);
  }

  // 7) tanh(c_t) -> f16
  {
    CvArgs a = {};
    a.src[0] = c_out; a.dst[0] = tanhc16; a.doTanh[0] = 1;
    conv_f32_to_f16_kernel<<<dim3(HN2 / (256 * 8), 1, 1), blk, 0, stream>>>(a);
  }

  // 8) h_t = tanh(c_t) @ o_t
  {
    Gemm2Args ga = {};
    ga.A1[0] = tanhc16; ga.B1[0] = o16t;
    ga.A2[0] = nullptr; ga.B2[0] = nullptr;
    ga.bias[0] = nullptr; ga.C[0] = h_out;
    gemm2_wmma_kernel<<<dim3(16, 16, 1), blk, 0, stream>>>(ga);
  }

  // 9) h_t -> f16
  {
    CvArgs a = {};
    a.src[0] = h_out; a.dst[0] = ht16;
    conv_f32_to_f16_kernel<<<dim3(HN2 / (256 * 8), 1, 1), blk, 0, stream>>>(a);
  }

  // 10) y_t = h_t @ W_y + b_y
  {
    Gemm2Args ga = {};
    ga.A1[0] = ht16; ga.B1[0] = wyt;
    ga.A2[0] = nullptr; ga.B2[0] = nullptr;
    ga.bias[0] = by; ga.C[0] = y_out;
    gemm2_wmma_kernel<<<dim3(16, 16, 1), blk, 0, stream>>>(ga);
  }
}
